// LamNet_50491635531985
// MI455X (gfx1250) — matrix operand order repaired and checked by reading, not compiled, not run on hardware
//
#include <hip/hip_runtime.h>
#include <math.h>

// ---------------- problem constants (from reference) ----------------
#define N_NODES   100000
#define HDIM      256
#define BGRAPH    512
#define NE_INTRA  800000
#define NE_INTER  400000
#define NH        (N_NODES * HDIM)          // 25,600,000
#define MTILES    (N_NODES / 16)            // 6250 (exact)
#define EPS_BN    1e-5f
#define SLOPE     0.01f

// ---------------- CDNA5 WMMA types ----------------
typedef __bf16 bf16;
typedef bf16        v16bf  __attribute__((ext_vector_type(16)));
typedef float       v8f    __attribute__((ext_vector_type(8)));
typedef unsigned int u32x4 __attribute__((ext_vector_type(4)));

union FragBF { v16bf v; u32x4 q[2]; };

__device__ __forceinline__ unsigned short f2bf(float f) {
  unsigned int u = __float_as_uint(f);
  u += 0x7FFFu + ((u >> 16) & 1u);   // round-to-nearest-even
  return (unsigned short)(u >> 16);
}
__device__ __forceinline__ float siluf(float x) { return x / (1.f + __expf(-x)); }
__device__ __forceinline__ float lrelu(float x) { return x > 0.f ? x : SLOPE * x; }

// ---------------- utility kernels ----------------
__global__ void zero_k(float* __restrict__ p, int n) {
  int i = blockIdx.x * blockDim.x + threadIdx.x;
  if (i < n) p[i] = 0.f;
}

// h = silu(x @ W + b); x:[N,35] W:[35,256]  — one block per node
__global__ void lin1_k(const float* __restrict__ x, const float* __restrict__ W,
                       const float* __restrict__ b, float* __restrict__ h) {
  int n = blockIdx.x, c = threadIdx.x;
  float s = b[c];
  const float* xr = x + n * 35;
#pragma unroll
  for (int k = 0; k < 35; ++k) s += xr[k] * W[k * HDIM + c];
  h[n * HDIM + c] = siluf(s);
}

// one wave32 per edge: d -> rbf(9) -> radial = silu(rbf@Wc+bc) -> atomic scatter
__global__ void edge_agg_k(const int* __restrict__ ei, int E,
                           const float* __restrict__ pos,
                           const float* __restrict__ h,
                           const float* __restrict__ Wc,
                           const float* __restrict__ bc,
                           float* __restrict__ out) {
  int e = blockIdx.x * 8 + (threadIdx.x >> 5);
  if (e >= E) return;
  int lane = threadIdx.x & 31;
  int row = ei[e], col = ei[E + e];
  float dx = pos[row * 3 + 0] - pos[col * 3 + 0];
  float dy = pos[row * 3 + 1] - pos[col * 3 + 1];
  float dz = pos[row * 3 + 2] - pos[col * 3 + 2];
  float d = sqrtf(dx * dx + dy * dy + dz * dz);
  float rbf[9];
#pragma unroll
  for (int k = 0; k < 9; ++k) {           // mu = 0.75k, 1/sigma = 1.5
    float t = (d - 0.75f * (float)k) * 1.5f;
    rbf[k] = __expf(-t * t);
  }
  const float* hr = h + (long)row * HDIM;
  float*       oc = out + (long)col * HDIM;
#pragma unroll
  for (int c = lane; c < HDIM; c += 32) {
    float r = bc[c];
#pragma unroll
    for (int k = 0; k < 9; ++k) r += rbf[k] * Wc[k * HDIM + c];
    atomicAdd(oc + c, hr[c] * siluf(r));
  }
}

// Abf = bf16(h + out)
__global__ void addcast_k(const float* __restrict__ a, const float* __restrict__ b,
                          unsigned short* __restrict__ o, int n) {
  int i = blockIdx.x * blockDim.x + threadIdx.x;
  if (i < n) o[i] = f2bf(a[i] + b[i]);
}

// WbfT[c*256+k] = bf16(W[k*256+c])   (transpose so B k-elements are contiguous)
__global__ void wconv_k(const float* __restrict__ W, unsigned short* __restrict__ Wt) {
  int id = blockIdx.x * blockDim.x + threadIdx.x;    // 65536
  int k = id >> 8, c = id & 255;
  Wt[c * HDIM + k] = f2bf(W[k * HDIM + c]);
}

// out[m,n] = leakyrelu( A(bf16)[m,:] . W[:,n] + bias[n] )  via v_wmma_f32_16x16x32_bf16.
// One wave computes a 16x64 output strip (4 accumulators): the A fragment is loaded
// once per K-step and reused by 4 back-to-back independent WMMAs (no D->A/B hazard,
// 4x less A traffic than tile-per-wave).
__global__ void gemm_bf16_k(const unsigned short* __restrict__ A,
                            const unsigned short* __restrict__ Wt,
                            const float* __restrict__ bias,
                            float* __restrict__ out, int mtiles) {
  int wid = (blockIdx.x * blockDim.x + threadIdx.x) >> 5;
  int tm = wid >> 2, tg = wid & 3;            // column group: cols [tg*64, tg*64+64)
  if (tm >= mtiles) return;                   // wave-uniform (EXEC stays all-ones)
  int l = threadIdx.x & 31;
  int lm = l & 15, hi = l >> 4;
  const unsigned short* Ab = A + (long)(tm * 16 + lm) * HDIM;
  const unsigned short* Bb0 = Wt + (long)(tg * 64 + 0  + lm) * HDIM;
  const unsigned short* Bb1 = Wt + (long)(tg * 64 + 16 + lm) * HDIM;
  const unsigned short* Bb2 = Wt + (long)(tg * 64 + 32 + lm) * HDIM;
  const unsigned short* Bb3 = Wt + (long)(tg * 64 + 48 + lm) * HDIM;
  __builtin_prefetch(Ab, 0, 0);
  v8f acc0 = {}, acc1 = {}, acc2 = {}, acc3 = {};
#pragma unroll
  for (int ks = 0; ks < 8; ++ks) {
    int k0 = ks * 32;
    FragBF fa, fb0, fb1, fb2, fb3;
    // A 16x32 bf16 layout: lane half 'hi' owns K = k0+16hi .. +15 (contiguous)
    fa.q[0] = *(const u32x4*)(Ab + k0 + hi * 16);
    fa.q[1] = *(const u32x4*)(Ab + k0 + hi * 16 + 8);
    // B 32x16 bf16 layout: lane half 'hi' owns K = k0+16hi .. +15 (contiguous in Wt)
    const u32x4* p0 = (const u32x4*)(Bb0 + k0 + hi * 16);
    const u32x4* p1 = (const u32x4*)(Bb1 + k0 + hi * 16);
    const u32x4* p2 = (const u32x4*)(Bb2 + k0 + hi * 16);
    const u32x4* p3 = (const u32x4*)(Bb3 + k0 + hi * 16);
    fb0.q[0] = p0[0]; fb0.q[1] = p0[1];
    fb1.q[0] = p1[0]; fb1.q[1] = p1[1];
    fb2.q[0] = p2[0]; fb2.q[1] = p2[1];
    fb3.q[0] = p3[0]; fb3.q[1] = p3[1];
    acc0 = __builtin_amdgcn_wmma_f32_16x16x32_bf16(false, fa.v, false, fb0.v,
                                                   (short)0, acc0, false, false);
    acc1 = __builtin_amdgcn_wmma_f32_16x16x32_bf16(false, fa.v, false, fb1.v,
                                                   (short)0, acc1, false, false);
    acc2 = __builtin_amdgcn_wmma_f32_16x16x32_bf16(false, fa.v, false, fb2.v,
                                                   (short)0, acc2, false, false);
    acc3 = __builtin_amdgcn_wmma_f32_16x16x32_bf16(false, fa.v, false, fb3.v,
                                                   (short)0, acc3, false, false);
  }
  // C/D striping: vgpr r holds row hi*8+r, lanes 0..15 are cols n (per 16x16 tile)
#pragma unroll
  for (int t = 0; t < 4; ++t) {
    v8f acc = (t == 0) ? acc0 : (t == 1) ? acc1 : (t == 2) ? acc2 : acc3;
    int n = tg * 64 + t * 16 + lm;
    float bn = bias[n];
#pragma unroll
    for (int r = 0; r < 8; ++r) {
      int m = tm * 16 + hi * 8 + r;
      out[(long)m * HDIM + n] = lrelu(acc[r] + bn);
    }
  }
}

// partial per-feature sums of ta, ta^2, tb, tb^2 into stats[4*256]
__global__ void stats_k(const float* __restrict__ ta, const float* __restrict__ tb,
                        float* __restrict__ stats, int nrows) {
  int f = threadIdx.x;
  int r0 = blockIdx.x * 512, r1 = min(r0 + 512, nrows);
  float sa = 0.f, qa = 0.f, sb = 0.f, qb = 0.f;
  for (int n = r0; n < r1; ++n) {
    float va = ta[(long)n * HDIM + f];
    float vb = tb[(long)n * HDIM + f];
    sa += va; qa += va * va; sb += vb; qb += vb * vb;
  }
  atomicAdd(&stats[f], sa);
  atomicAdd(&stats[256 + f], qa);
  atomicAdd(&stats[512 + f], sb);
  atomicAdd(&stats[768 + f], qb);
}

// h = bn(ta; ga,ba) + bn(tb; gb,bb)
__global__ void bn_apply_k(const float* __restrict__ ta, const float* __restrict__ tb,
                           const float* __restrict__ stats,
                           const float* __restrict__ ga, const float* __restrict__ ba,
                           const float* __restrict__ gb, const float* __restrict__ bb,
                           float* __restrict__ h, int nrows) {
  int i = blockIdx.x * blockDim.x + threadIdx.x;
  if (i >= nrows * HDIM) return;
  int f = i & 255;
  float inv = 1.f / (float)nrows;
  float mua = stats[f] * inv;
  float vara = stats[256 + f] * inv - mua * mua;
  float mub = stats[512 + f] * inv;
  float varb = stats[768 + f] * inv - mub * mub;
  float va = ga[f] * (ta[i] - mua) * rsqrtf(vara + EPS_BN) + ba[f];
  float vb = gb[f] * (tb[i] - mub) * rsqrtf(varb + EPS_BN) + bb[f];
  h[i] = va + vb;
}

// global add pool
__global__ void pool_k(const float* __restrict__ h, const int* __restrict__ batch,
                       float* __restrict__ g) {
  int i = blockIdx.x * blockDim.x + threadIdx.x;
  if (i >= NH) return;
  int n = i >> 8, f = i & 255;
  atomicAdd(&g[batch[n] * HDIM + f], h[i]);
}

// e2 = extra@upW; hcat = [g1*(1-e2), g2*e2, diff]
__global__ void mix_k(const float* __restrict__ extra, const float* __restrict__ upW,
                      const float* __restrict__ g1, const float* __restrict__ g2,
                      float* __restrict__ hcat) {
  int id = blockIdx.x * blockDim.x + threadIdx.x;   // 512*256
  int b = id >> 8, c = id & 255;
  float e2 = 0.f;
#pragma unroll
  for (int k = 0; k < 10; ++k) e2 += extra[b * 10 + k] * upW[k * HDIM + c];
  float v1 = g1[b * HDIM + c] * (1.f - e2);
  float v2 = g2[b * HDIM + c] * e2;
  hcat[b * 768 + c]        = v1;
  hcat[b * 768 + 256 + c]  = v2;
  hcat[b * 768 + 512 + c]  = v1 - v2;
}

// generic small dense: out[b,j] = act(in[b,:K] . W[:,j] + bias[j]), 512 rows
__global__ void dense_k(const float* __restrict__ in, const float* __restrict__ W,
                        const float* __restrict__ bias, float* __restrict__ out,
                        int K, int J, int act) {
  int id = blockIdx.x * blockDim.x + threadIdx.x;
  if (id >= BGRAPH * J) return;
  int b = id / J, j = id - b * J;
  float s = bias[j];
  for (int k = 0; k < K; ++k) s += in[b * K + k] * W[k * J + j];
  if (act == 1) s = fmaxf(s, 0.f);
  else if (act == 2) s = lrelu(s);
  out[b * J + j] = s;
}

// BN over 512 rows x 256 features, one block
__global__ void head_bn_k(const float* __restrict__ in, float* __restrict__ out,
                          const float* __restrict__ g, const float* __restrict__ beta) {
  int f = threadIdx.x;
  float s = 0.f, q = 0.f;
  for (int n = 0; n < BGRAPH; ++n) {
    float v = in[n * HDIM + f];
    s += v; q += v * v;
  }
  float mu = s / (float)BGRAPH;
  float var = q / (float)BGRAPH - mu * mu;
  float sc = g[f] * rsqrtf(var + EPS_BN);
  float bt = beta[f];
  for (int n = 0; n < BGRAPH; ++n)
    out[n * HDIM + f] = sc * (in[n * HDIM + f] - mu) + bt;
}

__global__ void out_k(const float* __restrict__ t, const float* __restrict__ W,
                      const float* __restrict__ b, float* __restrict__ y) {
  int bb = blockIdx.x * blockDim.x + threadIdx.x;
  if (bb >= BGRAPH) return;
  float s = b[0];
#pragma unroll 8
  for (int f = 0; f < HDIM; ++f) s += t[bb * HDIM + f] * W[f];
  y[bb] = s;
}

// ---------------- host orchestration ----------------
// input indices (setup_inputs insertion order, params flattened in sorted-key order):
// 0:x1 1:pos1 2:eii1 3:ein1 4:batch1 5:x2 6:pos2 7:eii2 8:ein2 9:batch2 10:extra
// 11..22: fc_hidden[i] {W,b,beta,g} x3
// 23:fc_out_W 24:fc_out_b
// 25..32: fnn (W,b) x4
// 33..68: hil[l] {coord_cov_W, coord_cov_b, coord_ncov_W, coord_ncov_b,
//                 cov_W, cov_b, cov_beta, cov_g, ncov_W, ncov_b, ncov_beta, ncov_g} x3
// 69:lin1_W 70:lin1_b 71:up_W

extern "C" void kernel_launch(void* const* d_in, const int* in_sizes, int n_in,
                              void* d_out, int out_size, void* d_ws, size_t ws_size,
                              hipStream_t stream) {
  (void)in_sizes; (void)n_in; (void)out_size; (void)ws_size;
  char* ws = (char*)d_ws;
  size_t off = 0;
  auto take = [&](size_t bytes) { char* p = ws + off; off += (bytes + 255) & ~(size_t)255; return p; };
  float*          h     = (float*)take((size_t)NH * 4);
  float*          bufA  = (float*)take((size_t)NH * 4);
  float*          bufB  = (float*)take((size_t)NH * 4);
  unsigned short* Abf   = (unsigned short*)take((size_t)NH * 2);
  unsigned short* WbfT  = (unsigned short*)take((size_t)HDIM * HDIM * 2);
  float*          stats = (float*)take(1024 * 4);
  float*          g1    = (float*)take((size_t)BGRAPH * HDIM * 4);
  float*          g2    = (float*)take((size_t)BGRAPH * HDIM * 4);
  float*          hcat  = (float*)take((size_t)BGRAPH * 768 * 4);
  float*          t1    = (float*)take((size_t)BGRAPH * 512 * 4);
  float*          t2    = (float*)take((size_t)BGRAPH * 512 * 4);

  const float* lin1W = (const float*)d_in[69];
  const float* lin1b = (const float*)d_in[70];

  const int ZB = NH / 256;                       // 100000 blocks
  const int GEMM_BLOCKS = (MTILES * 4) / 8;      // 25000 waves / 8 per block = 3125

  for (int br = 0; br < 2; ++br) {
    int base = br * 5;
    const float* x   = (const float*)d_in[base + 0];
    const float* pos = (const float*)d_in[base + 1];
    const int*   eii = (const int*)d_in[base + 2];
    const int*   ein = (const int*)d_in[base + 3];
    const int*   bat = (const int*)d_in[base + 4];

    lin1_k<<<N_NODES, HDIM, 0, stream>>>(x, lin1W, lin1b, h);

    for (int l = 0; l < 3; ++l) {
      int pb = 33 + l * 12;
      const float* ccW  = (const float*)d_in[pb + 0];
      const float* ccb  = (const float*)d_in[pb + 1];
      const float* cnW  = (const float*)d_in[pb + 2];
      const float* cnb  = (const float*)d_in[pb + 3];
      const float* covW = (const float*)d_in[pb + 4];
      const float* covb = (const float*)d_in[pb + 5];
      const float* covbeta = (const float*)d_in[pb + 6];
      const float* covg    = (const float*)d_in[pb + 7];
      const float* ncovW = (const float*)d_in[pb + 8];
      const float* ncovb = (const float*)d_in[pb + 9];
      const float* ncovbeta = (const float*)d_in[pb + 10];
      const float* ncovg    = (const float*)d_in[pb + 11];

      zero_k<<<ZB, 256, 0, stream>>>(bufA, NH);
      zero_k<<<ZB, 256, 0, stream>>>(bufB, NH);
      edge_agg_k<<<NE_INTRA / 8, 256, 0, stream>>>(eii, NE_INTRA, pos, h, ccW, ccb, bufA);
      edge_agg_k<<<NE_INTER / 8, 256, 0, stream>>>(ein, NE_INTER, pos, h, cnW, cnb, bufB);

      addcast_k<<<ZB, 256, 0, stream>>>(h, bufA, Abf, NH);
      wconv_k<<<256, 256, 0, stream>>>(covW, WbfT);
      gemm_bf16_k<<<GEMM_BLOCKS, 256, 0, stream>>>(Abf, WbfT, covb, bufA, MTILES);

      addcast_k<<<ZB, 256, 0, stream>>>(h, bufB, Abf, NH);
      wconv_k<<<256, 256, 0, stream>>>(ncovW, WbfT);
      gemm_bf16_k<<<GEMM_BLOCKS, 256, 0, stream>>>(Abf, WbfT, ncovb, bufB, MTILES);

      zero_k<<<4, 256, 0, stream>>>(stats, 1024);
      stats_k<<<(N_NODES + 511) / 512, 256, 0, stream>>>(bufA, bufB, stats, N_NODES);
      bn_apply_k<<<ZB, 256, 0, stream>>>(bufA, bufB, stats,
                                         covg, covbeta, ncovg, ncovbeta, h, N_NODES);
    }

    float* g = (br == 0) ? g1 : g2;
    zero_k<<<(BGRAPH * HDIM) / 256, 256, 0, stream>>>(g, BGRAPH * HDIM);
    pool_k<<<ZB, 256, 0, stream>>>(h, bat, g);
  }

  // head
  const float* extra = (const float*)d_in[10];
  const float* upW   = (const float*)d_in[71];
  mix_k<<<(BGRAPH * HDIM) / 256, 256, 0, stream>>>(extra, upW, g1, g2, hcat);

  dense_k<<<(BGRAPH * 512) / 256, 256, 0, stream>>>(hcat, (const float*)d_in[25],
      (const float*)d_in[26], t1, 768, 512, 1);
  dense_k<<<(BGRAPH * 256) / 256, 256, 0, stream>>>(t1, (const float*)d_in[27],
      (const float*)d_in[28], t2, 512, 256, 1);
  dense_k<<<(BGRAPH * 128) / 256, 256, 0, stream>>>(t2, (const float*)d_in[29],
      (const float*)d_in[30], t1, 256, 128, 1);
  dense_k<<<(BGRAPH * 256) / 256, 256, 0, stream>>>(t1, (const float*)d_in[31],
      (const float*)d_in[32], t2, 128, 256, 0);

  for (int i = 0; i < 3; ++i) {
    int fb = 11 + 4 * i;
    dense_k<<<(BGRAPH * 256) / 256, 256, 0, stream>>>(t2, (const float*)d_in[fb],
        (const float*)d_in[fb + 1], t1, 256, 256, 2);
    head_bn_k<<<1, 256, 0, stream>>>(t1, t2,
        (const float*)d_in[fb + 3], (const float*)d_in[fb + 2]);
  }

  out_k<<<2, 256, 0, stream>>>(t2, (const float*)d_in[23], (const float*)d_in[24],
                               (float*)d_out);
}
